// Model_2104533975619
// MI455X (gfx1250) — matrix operand order repaired and checked by reading, not compiled, not run on hardware
//
#include <hip/hip_runtime.h>
#include <hip/hip_bf16.h>

typedef __bf16 bf16_t;
typedef __attribute__((ext_vector_type(16))) __bf16 v16bf;
typedef __attribute__((ext_vector_type(8)))  __bf16 v8bf;
typedef __attribute__((ext_vector_type(8)))  float  v8f;

#define BB 32
#define NN 4096
#define SS 256
#define DD 256
#define TOPK 100

#define MODE_LN_RELU 0
#define MODE_BIAS    1
#define MODE_BIAS_T  2

__device__ __forceinline__ v8f vzero8() {
  v8f z;
#pragma unroll
  for (int i = 0; i < 8; ++i) z[i] = 0.0f;
  return z;
}

// reductions across the 16-lane half (lanes differ in bits [3:0]) — wave32
__device__ __forceinline__ float half_sum(float v) {
#pragma unroll
  for (int m = 1; m <= 8; m <<= 1) v += __shfl_xor(v, m, 32);
  return v;
}
__device__ __forceinline__ float half_max(float v) {
#pragma unroll
  for (int m = 1; m <= 8; m <<= 1) v = fmaxf(v, __shfl_xor(v, m, 32));
  return v;
}
__device__ __forceinline__ float wave_sum(float v) {
#pragma unroll
  for (int m = 1; m <= 16; m <<= 1) v += __shfl_xor(v, m, 32);
  return v;
}

// A fragment: 16x32 bf16, M = lane&15; lanes<16 hold K = {k0..k0+7, k0+16..k0+23},
// lanes>=16 hold K+8 (ISA 7.12.2, 16-bit A 16x32). Two contiguous b128 loads.
__device__ __forceinline__ v16bf load_a_frag(const bf16_t* base, int ld, int row0, int k0) {
  const int lane = threadIdx.x & 31;
  const bf16_t* p = base + (size_t)(row0 + (lane & 15)) * ld + k0 + ((lane >> 4) << 3);
  v8bf lo = *(const v8bf*)p;
  v8bf hi = *(const v8bf*)(p + 16);
  v16bf f;
#pragma unroll
  for (int e = 0; e < 8; ++e) { f[e] = lo[e]; f[e + 8] = hi[e]; }
  return f;
}

// B fragment: 32x16 bf16 supplied via B^T [N,K] row-major. N = lane&15,
// lanes<16 hold K = {k0..k0+7, k0+16..k0+23}, lanes>=16 hold K+8 (matches A).
__device__ __forceinline__ v16bf load_b_frag(const bf16_t* baseT, int ld, int n0, int k0) {
  const int lane = threadIdx.x & 31;
  const bf16_t* p = baseT + (size_t)(n0 + (lane & 15)) * ld + k0 + ((lane >> 4) << 3);
  v8bf lo = *(const v8bf*)p;
  v8bf hi = *(const v8bf*)(p + 16);
  v16bf f;
#pragma unroll
  for (int e = 0; e < 8; ++e) { f[e] = lo[e]; f[e + 8] = hi[e]; }
  return f;
}

__device__ __forceinline__ v8f wmma_bf16f32(v16bf a, v16bf b, v8f c) {
  return __builtin_amdgcn_wmma_f32_16x16x32_bf16(false, a, false, b, (short)0, c, false, false);
}

// ---------------------------------------------------------------- prep kernels
__global__ void transpose_to_bf16_kernel(const float* __restrict__ W, bf16_t* __restrict__ Wt,
                                         int K, int Nout) {
  int idx = blockIdx.x * 256 + threadIdx.x;
  if (idx >= K * Nout) return;
  int o = idx / K, i = idx - o * K;           // Wt[o][i] = W[i][o]
  Wt[(size_t)o * K + i] = (bf16_t)W[(size_t)i * Nout + o];
}

__global__ void to_bf16_kernel(const float* __restrict__ in, bf16_t* __restrict__ out, int n) {
  int idx = blockIdx.x * 256 + threadIdx.x;
  if (idx < n) out[idx] = (bf16_t)in[idx];
}

// ------------------------------------------------- encoder layer 1 (K=3, VALU)
__global__ void enc1_kernel(const float* __restrict__ tc, const float* __restrict__ W1,
                            const float* __restrict__ b1, const float* __restrict__ g1,
                            const float* __restrict__ e1, bf16_t* __restrict__ c_out, int rows) {
  int wave = threadIdx.x >> 5, lane = threadIdx.x & 31;
  int row = blockIdx.x * 8 + wave;
  if (row >= rows) return;
  float x0 = tc[(size_t)row * 3 + 0], x1 = tc[(size_t)row * 3 + 1], x2 = tc[(size_t)row * 3 + 2];
  int c0 = lane * 8;
  float v[8], s = 0.f, sq = 0.f;
#pragma unroll
  for (int j = 0; j < 8; ++j) {
    int col = c0 + j;
    float y = b1[col] + x0 * W1[col] + x1 * W1[256 + col] + x2 * W1[512 + col];
    v[j] = y; s += y; sq += y * y;
  }
  s = wave_sum(s); sq = wave_sum(sq);
  float mean = s * (1.f / 256.f);
  float var  = sq * (1.f / 256.f) - mean * mean;
  float rstd = rsqrtf(var + 1e-5f);
#pragma unroll
  for (int j = 0; j < 8; ++j) {
    int col = c0 + j;
    float y = (v[j] - mean) * rstd * g1[col] + e1[col];
    c_out[(size_t)row * 256 + col] = (bf16_t)fmaxf(y, 0.f);
  }
}

// -------------------- generic rows x 256 GEMM (+bias [+LN+ReLU] / transposed)
// A = [A1 | A2] (concat along K), Kdim in {256,512}. 4 accumulator tiles live;
// all fragment loads of a k-step are issued before the WMMA burst so the 5
// loads pipeline and overlap the XDL pipe across iterations.
__global__ void gemm256_kernel(const bf16_t* __restrict__ A1, const bf16_t* __restrict__ A2,
                               int lda, const bf16_t* __restrict__ Wt, int Kdim,
                               const float* __restrict__ bias, const float* __restrict__ gamma,
                               const float* __restrict__ beta, bf16_t* __restrict__ out,
                               int rows, int mode, int Sper) {
  extern __shared__ char smem_raw[];
  int tid = threadIdx.x, wave = tid >> 5, lane = tid & 31;
  float* yw = (float*)smem_raw + (size_t)wave * 16 * 256;   // 16 KB per wave
  int row0 = blockIdx.x * 128 + wave * 16;
  if (row0 >= rows) return;
  const int nidx = lane & 15;
  const int mb   = (lane >> 4) << 3;
  const int kt   = Kdim >> 5;

  float s[8], sq[8];
#pragma unroll
  for (int i = 0; i < 8; ++i) { s[i] = 0.f; sq[i] = 0.f; }

  for (int g = 0; g < 4; ++g) {                 // 4 column groups of 64
    v8f acc[4];
#pragma unroll
    for (int t = 0; t < 4; ++t) acc[t] = vzero8();
    for (int kc = 0; kc < kt; ++kc) {
      int k0 = kc << 5;
      const bf16_t* Ap = A1; int ka = k0;
      if (A2 != nullptr && k0 >= 256) { Ap = A2; ka = k0 - 256; }
      v16bf a = load_a_frag(Ap, lda, row0, ka);
      v16bf bfr[4];
#pragma unroll
      for (int t = 0; t < 4; ++t)
        bfr[t] = load_b_frag(Wt, Kdim, ((g << 2) + t) << 4, k0);
#pragma unroll
      for (int t = 0; t < 4; ++t)
        acc[t] = wmma_bf16f32(a, bfr[t], acc[t]);   // v_wmma_f32_16x16x32_bf16
    }
#pragma unroll
    for (int t = 0; t < 4; ++t) {
      int col = (((g << 2) + t) << 4) + nidx;
      float bv = bias[col];
      if (mode == MODE_LN_RELU) {
#pragma unroll
        for (int i = 0; i < 8; ++i) {
          float y = acc[t][i] + bv;
          yw[(mb + i) * 256 + col] = y;
          s[i] += y; sq[i] += y * y;
        }
      } else if (mode == MODE_BIAS) {
#pragma unroll
        for (int i = 0; i < 8; ++i)
          out[(size_t)(row0 + mb + i) * 256 + col] = (bf16_t)(acc[t][i] + bv);
      } else {                                   // transposed store [b][col][s]
#pragma unroll
        for (int i = 0; i < 8; ++i) {
          int row = row0 + mb + i;
          int bb = row / Sper, sidx = row - bb * Sper;
          out[((size_t)bb * 256 + col) * Sper + sidx] = (bf16_t)(acc[t][i] + bv);
        }
      }
    }
  }
  if (mode != MODE_LN_RELU) return;

  float mean[8], rstd[8];
#pragma unroll
  for (int i = 0; i < 8; ++i) {
    float ts = half_sum(s[i]), tq = half_sum(sq[i]);
    mean[i] = ts * (1.f / 256.f);
    float var = tq * (1.f / 256.f) - mean[i] * mean[i];
    rstd[i] = rsqrtf(var + 1e-5f);
  }
#pragma unroll
  for (int t = 0; t < 16; ++t) {
    int col = (t << 4) + nidx;
    float g = gamma[col], be = beta[col];
#pragma unroll
    for (int i = 0; i < 8; ++i) {
      float y = (yw[(mb + i) * 256 + col] - mean[i]) * rstd[i] * g + be;
      out[(size_t)(row0 + mb + i) * 256 + col] = (bf16_t)fmaxf(y, 0.f);
    }
  }
}

// -------------------------- fused attention: softmax(q k^T / 16) v  per block
__global__ void attn_kernel(const bf16_t* __restrict__ q, const bf16_t* __restrict__ k,
                            const bf16_t* __restrict__ vT, bf16_t* __restrict__ hout) {
  extern __shared__ char smem_raw[];
  bf16_t* kl = (bf16_t*)smem_raw;                       // k[b]: [SS, DD] bf16 (128 KB)
  int b = blockIdx.y;
  int tid = threadIdx.x, wave = tid >> 5, lane = tid & 31;
  const bf16_t* kbase = k + (size_t)b * SS * DD;
  int nch = (SS * DD) >> 3;
  for (int i = tid; i < nch; i += blockDim.x) ((v8bf*)kl)[i] = ((const v8bf*)kbase)[i];
  __syncthreads();

  bf16_t* Pm = (bf16_t*)(smem_raw + (size_t)SS * DD * 2) + (size_t)wave * 16 * SS;  // 8 KB/wave
  int row0 = b * NN + blockIdx.x * 128 + wave * 16;
  const int nidx = lane & 15, mb = (lane >> 4) << 3;
  const float scale = 0.0625f;                           // 1/sqrt(256)

  // phase 1: scores, 4 s-tile groups, raw scaled scores -> Pm (bf16), max in regs
  float mx[8];
#pragma unroll
  for (int i = 0; i < 8; ++i) mx[i] = -3.0e38f;
  for (int g = 0; g < 4; ++g) {
    v8f acc[4];
#pragma unroll
    for (int t = 0; t < 4; ++t) acc[t] = vzero8();
    for (int kc = 0; kc < DD / 32; ++kc) {
      v16bf a = load_a_frag(q, DD, row0, kc << 5);
      v16bf bfr[4];
#pragma unroll
      for (int t = 0; t < 4; ++t)
        bfr[t] = load_b_frag(kl, DD, (((g << 2) + t) << 4), kc << 5);
#pragma unroll
      for (int t = 0; t < 4; ++t)
        acc[t] = wmma_bf16f32(a, bfr[t], acc[t]);
    }
#pragma unroll
    for (int t = 0; t < 4; ++t) {
      int sc0 = ((((g << 2) + t) << 4)) + nidx;
#pragma unroll
      for (int i = 0; i < 8; ++i) {
        float y = acc[t][i] * scale;
        mx[i] = fmaxf(mx[i], y);
        Pm[(mb + i) * SS + sc0] = (bf16_t)y;
      }
    }
  }
#pragma unroll
  for (int i = 0; i < 8; ++i) mx[i] = half_max(mx[i]);

  // softmax sweeps over LDS
  float sum[8];
#pragma unroll
  for (int i = 0; i < 8; ++i) sum[i] = 0.f;
#pragma unroll
  for (int t = 0; t < 16; ++t) {
    int sc0 = (t << 4) + nidx;
#pragma unroll
    for (int i = 0; i < 8; ++i) {
      float e = __expf((float)Pm[(mb + i) * SS + sc0] - mx[i]);
      sum[i] += e;
      Pm[(mb + i) * SS + sc0] = (bf16_t)e;
    }
  }
#pragma unroll
  for (int i = 0; i < 8; ++i) sum[i] = 1.f / half_sum(sum[i]);
#pragma unroll
  for (int t = 0; t < 16; ++t) {
    int sc0 = (t << 4) + nidx;
#pragma unroll
    for (int i = 0; i < 8; ++i)
      Pm[(mb + i) * SS + sc0] = (bf16_t)((float)Pm[(mb + i) * SS + sc0] * sum[i]);
  }

  // phase 2: h = P @ v via vT [DD, SS]; 4-tile groups, one A-frag feeds 4 WMMAs
  const bf16_t* vTb = vT + (size_t)b * DD * SS;
  for (int g = 0; g < 4; ++g) {
    v8f acc[4];
#pragma unroll
    for (int t = 0; t < 4; ++t) acc[t] = vzero8();
    for (int kc = 0; kc < SS / 32; ++kc) {
      v16bf a = load_a_frag(Pm, SS, 0, kc << 5);
      v16bf bfr[4];
#pragma unroll
      for (int t = 0; t < 4; ++t)
        bfr[t] = load_b_frag(vTb, SS, (((g << 2) + t) << 4), kc << 5);
#pragma unroll
      for (int t = 0; t < 4; ++t)
        acc[t] = wmma_bf16f32(a, bfr[t], acc[t]);
    }
#pragma unroll
    for (int t = 0; t < 4; ++t) {
      int col = (((g << 2) + t) << 4) + nidx;
#pragma unroll
      for (int i = 0; i < 8; ++i)
        hout[(size_t)(row0 + mb + i) * 256 + col] = (bf16_t)acc[t][i];
    }
  }
}

// ------------------------- scores = [c,h,m] . Wd2 + bd2, masked, one wave/row
__global__ void score_kernel(const bf16_t* __restrict__ c, const bf16_t* __restrict__ h,
                             const bf16_t* __restrict__ m, const float* __restrict__ Wd2,
                             const float* __restrict__ bd2, const int* __restrict__ mask,
                             float* __restrict__ scores, int rows) {
  int wave = threadIdx.x >> 5, lane = threadIdx.x & 31;
  int row = blockIdx.x * 8 + wave;
  if (row >= rows) return;
  float s = 0.f;
  int c0 = lane * 8;
#pragma unroll
  for (int j = 0; j < 8; ++j) {
    int col = c0 + j;
    size_t o = (size_t)row * 256 + col;
    s += (float)c[o] * Wd2[col] + (float)h[o] * Wd2[256 + col] + (float)m[o] * Wd2[512 + col];
  }
  s = wave_sum(s);
  if (lane == 0) scores[row] = mask[row] ? (s + bd2[0]) : -1e9f;
}

// ---------------- per-batch: softmax, top-100, softmax over top vals, scatter
__global__ void topk_kernel(const float* __restrict__ scores, float* __restrict__ prob) {
  __shared__ float sv[NN];
  __shared__ float rv[256];
  __shared__ int   ri[256];
  __shared__ float topv[TOPK];
  __shared__ int   topi[TOPK];
  int b = blockIdx.x, tid = threadIdx.x;
  const float* sr = scores + (size_t)b * NN;
  float* pr = prob + (size_t)b * NN;

  float lmax = -3.0e38f;
  for (int i = tid; i < NN; i += 256) { float v = sr[i]; sv[i] = v; pr[i] = 0.f; lmax = fmaxf(lmax, v); }
  rv[tid] = lmax; __syncthreads();
  for (int off = 128; off > 0; off >>= 1) { if (tid < off) rv[tid] = fmaxf(rv[tid], rv[tid + off]); __syncthreads(); }
  float bmax = rv[0]; __syncthreads();
  float lsum = 0.f;
  for (int i = tid; i < NN; i += 256) { float e = __expf(sv[i] - bmax); sv[i] = e; lsum += e; }
  rv[tid] = lsum; __syncthreads();
  for (int off = 128; off > 0; off >>= 1) { if (tid < off) rv[tid] += rv[tid + off]; __syncthreads(); }
  float inv = 1.f / rv[0]; __syncthreads();
  for (int i = tid; i < NN; i += 256) sv[i] *= inv;
  __syncthreads();

  for (int kk = 0; kk < TOPK; ++kk) {
    float bv = -3.0e38f; int bi = NN;
    for (int i = tid; i < NN; i += 256) { float v = sv[i]; if (v > bv) { bv = v; bi = i; } }
    rv[tid] = bv; ri[tid] = bi; __syncthreads();
    for (int off = 128; off > 0; off >>= 1) {
      if (tid < off) {
        if (rv[tid + off] > rv[tid] || (rv[tid + off] == rv[tid] && ri[tid + off] < ri[tid])) {
          rv[tid] = rv[tid + off]; ri[tid] = ri[tid + off];
        }
      }
      __syncthreads();
    }
    if (tid == 0) { topv[kk] = rv[0]; topi[kk] = ri[0]; sv[ri[0]] = -1.f; }
    __syncthreads();
  }

  __threadfence();
  if (tid == 0) {
    float m = -3.0e38f;
    for (int kk = 0; kk < TOPK; ++kk) m = fmaxf(m, topv[kk]);
    float s = 0.f;
    for (int kk = 0; kk < TOPK; ++kk) { float e = __expf(topv[kk] - m); topv[kk] = e; s += e; }
    float is = 1.f / s;
    for (int kk = 0; kk < TOPK; ++kk) pr[topi[kk]] = topv[kk] * is;
  }
}

// ----------------------------------------------------------------------------
extern "C" void kernel_launch(void* const* d_in, const int* in_sizes, int n_in,
                              void* d_out, int out_size, void* d_ws, size_t ws_size,
                              hipStream_t stream) {
  (void)in_sizes; (void)n_in; (void)out_size; (void)ws_size;
  const float* state = (const float*)d_in[0];
  const float* tcand = (const float*)d_in[1];
  const int*   mask  = (const int*)d_in[2];
  const float* W1 = (const float*)d_in[3];  const float* b1 = (const float*)d_in[4];
  const float* g1 = (const float*)d_in[5];  const float* e1 = (const float*)d_in[6];
  const float* W2 = (const float*)d_in[7];  const float* b2 = (const float*)d_in[8];
  const float* g2 = (const float*)d_in[9];  const float* e2 = (const float*)d_in[10];
  const float* W3 = (const float*)d_in[11]; const float* b3 = (const float*)d_in[12];
  const float* g3 = (const float*)d_in[13]; const float* e3 = (const float*)d_in[14];
  const float* Wq = (const float*)d_in[15]; const float* bq = (const float*)d_in[16];
  const float* Wk = (const float*)d_in[17]; const float* bk = (const float*)d_in[18];
  const float* Wv = (const float*)d_in[19]; const float* bv = (const float*)d_in[20];
  const float* Wd1 = (const float*)d_in[21]; const float* bd1 = (const float*)d_in[22];
  const float* gd  = (const float*)d_in[23]; const float* ed  = (const float*)d_in[24];
  const float* Wd2 = (const float*)d_in[25]; const float* bd2 = (const float*)d_in[26];

  char* ws = (char*)d_ws;
  size_t off = 0;
  auto alloc = [&](size_t bytes) -> void* {
    void* p = ws + off;
    off = (off + bytes + 255) & ~(size_t)255;
    return p;
  };
  const size_t rowsBN = (size_t)BB * NN, rowsBS = (size_t)BB * SS;
  bf16_t* W2t  = (bf16_t*)alloc(256 * 256 * 2);
  bf16_t* W3t  = (bf16_t*)alloc(256 * 256 * 2);
  bf16_t* Wqt  = (bf16_t*)alloc(256 * 256 * 2);
  bf16_t* Wkt  = (bf16_t*)alloc(256 * 256 * 2);
  bf16_t* Wvt  = (bf16_t*)alloc(256 * 256 * 2);
  bf16_t* Wd1t = (bf16_t*)alloc(512 * 256 * 2);
  bf16_t* stb  = (bf16_t*)alloc(rowsBS * 256 * 2);
  bf16_t* cA   = (bf16_t*)alloc(rowsBN * 256 * 2);   // c (final encoder output)
  bf16_t* cB   = (bf16_t*)alloc(rowsBN * 256 * 2);   // enc tmp, then h
  bf16_t* qb   = (bf16_t*)alloc(rowsBN * 256 * 2);   // q, then decoder MLP out
  bf16_t* kb   = (bf16_t*)alloc(rowsBS * 256 * 2);
  bf16_t* vTb  = (bf16_t*)alloc(rowsBS * 256 * 2);
  float*  sc   = (float*)alloc(rowsBN * 4);

  transpose_to_bf16_kernel<<<256, 256, 0, stream>>>(W2, W2t, 256, 256);
  transpose_to_bf16_kernel<<<256, 256, 0, stream>>>(W3, W3t, 256, 256);
  transpose_to_bf16_kernel<<<256, 256, 0, stream>>>(Wq, Wqt, 256, 256);
  transpose_to_bf16_kernel<<<256, 256, 0, stream>>>(Wk, Wkt, 256, 256);
  transpose_to_bf16_kernel<<<256, 256, 0, stream>>>(Wv, Wvt, 256, 256);
  transpose_to_bf16_kernel<<<512, 256, 0, stream>>>(Wd1, Wd1t, 512, 256);
  to_bf16_kernel<<<(int)((rowsBS * 256 + 255) / 256), 256, 0, stream>>>(state, stb, (int)(rowsBS * 256));

  enc1_kernel<<<(int)(rowsBN / 8), 256, 0, stream>>>(tcand, W1, b1, g1, e1, cA, (int)rowsBN);

  const size_t shGemm = 8 * 16 * 256 * 4;             // 128 KB per-wave f32 staging
  gemm256_kernel<<<(int)(rowsBN / 128), 256, shGemm, stream>>>(cA, nullptr, 256, W2t, 256, b2, g2, e2, cB, (int)rowsBN, MODE_LN_RELU, 0);
  gemm256_kernel<<<(int)(rowsBN / 128), 256, shGemm, stream>>>(cB, nullptr, 256, W3t, 256, b3, g3, e3, cA, (int)rowsBN, MODE_LN_RELU, 0);
  gemm256_kernel<<<(int)(rowsBN / 128), 256, shGemm, stream>>>(cA, nullptr, 256, Wqt, 256, bq, nullptr, nullptr, qb, (int)rowsBN, MODE_BIAS, 0);
  gemm256_kernel<<<(int)(rowsBS / 128), 256, shGemm, stream>>>(stb, nullptr, 256, Wkt, 256, bk, nullptr, nullptr, kb, (int)rowsBS, MODE_BIAS, 0);
  gemm256_kernel<<<(int)(rowsBS / 128), 256, shGemm, stream>>>(stb, nullptr, 256, Wvt, 256, bv, nullptr, nullptr, vTb, (int)rowsBS, MODE_BIAS_T, SS);

  const size_t shAttn = (size_t)SS * DD * 2 + 8 * 16 * SS * 2;  // 128 KB k + 64 KB P
  attn_kernel<<<dim3(NN / 128, BB), 256, shAttn, stream>>>(qb, kb, vTb, cB);

  gemm256_kernel<<<(int)(rowsBN / 128), 256, shGemm, stream>>>(cA, cB, 256, Wd1t, 512, bd1, gd, ed, qb, (int)rowsBN, MODE_LN_RELU, 0);

  score_kernel<<<(int)(rowsBN / 8), 256, 0, stream>>>(cA, cB, qb, Wd2, bd2, mask, sc, (int)rowsBN);
  topk_kernel<<<BB, 256, 0, stream>>>(sc, (float*)d_out);
}
